// CombinedNN_65635690217686
// MI455X (gfx1250) — matrix operand order repaired and checked by reading, not compiled, not run on hardware
//
#include <hip/hip_runtime.h>
#include <math.h>

#define LNUM 2
#define BNUM 2
#define SNUM 1024
#define DNUM 512
#define HNUM 32
#define FNUM 2048
#define CNUM 1000

typedef __bf16 bf16_t;
typedef __bf16 bf16x16 __attribute__((ext_vector_type(16)));
typedef float  floatx8 __attribute__((ext_vector_type(8)));

__device__ __forceinline__ unsigned short f32_to_bf16_bits(float f) {
  unsigned int u = __builtin_bit_cast(unsigned int, f);
  u += 0x7FFFu + ((u >> 16) & 1u);   // round-to-nearest-even
  return (unsigned short)(u >> 16);
}

// LDS byte offset of a __shared__ object: generic LDS addresses carry the
// LDS offset in addr[31:0] (CDNA5 aperture mapping), so truncation is exact.
__device__ __forceinline__ unsigned int lds_off(const void* p) {
  return (unsigned int)(unsigned long long)(uintptr_t)p;
}

__device__ __forceinline__ void async_b128(unsigned int lds_addr, const void* gaddr) {
  asm volatile("global_load_async_to_lds_b128 %0, %1, off"
               :: "v"(lds_addr), "v"((unsigned long long)(uintptr_t)gaddr)
               : "memory");
}
__device__ __forceinline__ void wait_async0() {
  asm volatile("s_wait_asynccnt 0x0" ::: "memory");
}

// ---------------------------------------------------------------- convert
__global__ void cvt_f32_bf16_kernel(const float* __restrict__ src,
                                    unsigned short* __restrict__ dst, int n) {
  int i = blockIdx.x * blockDim.x + threadIdx.x;
  int stride = gridDim.x * blockDim.x;
  for (; i < n; i += stride) dst[i] = f32_to_bf16_bits(src[i]);
}

// src: [L][K][N] fp32 row-major -> dst: [L][N][K] bf16 (transposed).
// Grid: (N/32, K/32, L), block 256 (32x8).
__global__ __launch_bounds__(256)
void cvt_transpose_bf16_kernel(const float* __restrict__ src,
                               unsigned short* __restrict__ dst, int K, int N)
{
  __shared__ float tile[32][33];
  const int kb = blockIdx.y * 32;
  const int nb = blockIdx.x * 32;
  const float* s = src + (size_t)blockIdx.z * K * N;
  unsigned short* d = dst + (size_t)blockIdx.z * K * N;
  const int tx = threadIdx.x & 31, ty = threadIdx.x >> 5;
  #pragma unroll
  for (int q = 0; q < 4; ++q)
    tile[ty + q * 8][tx] = s[(size_t)(kb + ty + q * 8) * N + nb + tx];
  __syncthreads();
  #pragma unroll
  for (int q = 0; q < 4; ++q)
    d[(size_t)(nb + ty + q * 8) * K + kb + tx] =
        f32_to_bf16_bits(tile[tx][ty + q * 8]);
}

// ---------------------------------------------------------------- WMMA GEMM
// C[M,N] = scale * A[M,K] * B^T (+ bias[n]) (+ relu), B stored [N,K] (K-major).
// OUTMODE: 0 = fp32 C[M,N]; 1 = bf16 C[M,N]; 2 = bf16 transposed C^T[N,M]
// (ldc applies to the stored matrix: mode 0/1 row stride of C, mode 2 row
//  stride of C^T). 256 threads = 8 waves; block tile 128(M) x 64(N); each
// wave one 32x32 tile = 2x2 WMMA 16x16x32 bf16. Double-buffered LDS staged
// with GLOBAL_LOAD_ASYNC_TO_LDS_B128 (ASYNCcnt), one barrier per K-step.
// Requires M%128==0, N%64==0, K%32==0 (true at all call sites).
template<int OUTMODE, int RELU>
__global__ __launch_bounds__(256)
void gemm_wmma_kernel(const unsigned short* __restrict__ A,
                      const unsigned short* __restrict__ Bm,
                      void* __restrict__ Cv,
                      const float* __restrict__ bias,
                      float scale, int M, int N, int K,
                      int lda, int ldb, int ldc,
                      long long aStride, long long bStride, long long cStride)
{
  // rows padded to 40 ushort (80B = 16B-multiple; frag dwords stay 4B-aligned)
  __shared__ unsigned short As[2][128 * 40];   // A tile: 128 rows x 32 k
  __shared__ unsigned short Bs[2][64 * 40];    // B tile: 64 n-rows x 32 k

  const int t    = threadIdx.x;
  const int wave = t >> 5;
  const int lane = t & 31;
  const int half = lane >> 4;
  const int id   = lane & 15;
  const int wm   = (wave & 3) * 32;   // wave row group (32 rows)
  const int wn   = (wave >> 2) * 32;  // wave col group (32 cols)

  const int m0 = blockIdx.y * 128;
  const int n0 = blockIdx.x * 64;
  const long long z = blockIdx.z;

  const unsigned short* Ab = A  + (size_t)z * aStride;
  const unsigned short* Bb = Bm + (size_t)z * bStride;

  floatx8 acc00 = {0,0,0,0,0,0,0,0}, acc01 = acc00, acc10 = acc00, acc11 = acc00;

  // async staging assignments (16B per lane per op)
  const int ar0 = t >> 2;            // A rows t>>2 and t>>2 + 64
  const int ac  = (t & 3) * 8;       // A k-chunk (8 ushorts)
  const int bnr = t >> 2;            // B n-row 0..63
  const int bkc = (t & 3) * 8;       // B k-chunk

  auto issueTile = [&](int buf, int k0) {
    async_b128(lds_off(&As[buf][ar0 * 40 + ac]),
               Ab + (size_t)(m0 + ar0) * lda + k0 + ac);
    async_b128(lds_off(&As[buf][(ar0 + 64) * 40 + ac]),
               Ab + (size_t)(m0 + ar0 + 64) * lda + k0 + ac);
    async_b128(lds_off(&Bs[buf][bnr * 40 + bkc]),
               Bb + (size_t)(n0 + bnr) * ldb + k0 + bkc);
  };

  const int nsteps = K >> 5;
  issueTile(0, 0);

  union FragU { bf16x16 v; unsigned int u[8]; };

  for (int i = 0; i < nsteps; ++i) {
    wait_async0();          // my lanes' staged data has landed
    __syncthreads();        // everyone else's too; prev buffer free to reuse
    const int cur = i & 1;
    if (i + 1 < nsteps) issueTile(cur ^ 1, (i + 1) * 32);

    FragU a0, a1, b0, b1;
    #pragma unroll
    for (int r = 0; r < 8; ++r) {
      // A 16x32 fragment: lanes0-15 K 0..7 (v0-3) / 16..23 (v4-7); lanes16-31 +8
      const int ka = ((r < 4) ? (2 * r) : (16 + 2 * (r - 4))) + 8 * half;
      a0.u[r] = *reinterpret_cast<const unsigned int*>(&As[cur][(wm + id) * 40 + ka]);
      a1.u[r] = *reinterpret_cast<const unsigned int*>(&As[cur][(wm + 16 + id) * 40 + ka]);
      // B 32x16 fragment: lanes0-15 K 0..7 (v0-3) / 16..23 (v4-7); lanes16-31 +8; col = lane%16
      const int kb = ((r < 4) ? (2 * r) : (16 + 2 * (r - 4))) + 8 * half;
      b0.u[r] = *reinterpret_cast<const unsigned int*>(&Bs[cur][(wn + id) * 40 + kb]);
      b1.u[r] = *reinterpret_cast<const unsigned int*>(&Bs[cur][(wn + 16 + id) * 40 + kb]);
    }
    acc00 = __builtin_amdgcn_wmma_f32_16x16x32_bf16(false, a0.v, false, b0.v, (short)0, acc00, false, false);
    acc01 = __builtin_amdgcn_wmma_f32_16x16x32_bf16(false, a0.v, false, b1.v, (short)0, acc01, false, false);
    acc10 = __builtin_amdgcn_wmma_f32_16x16x32_bf16(false, a1.v, false, b0.v, (short)0, acc10, false, false);
    acc11 = __builtin_amdgcn_wmma_f32_16x16x32_bf16(false, a1.v, false, b1.v, (short)0, acc11, false, false);
  }

  // Epilogue. D layout: VGPR r -> row r (+8 for lanes 16-31), col = lane%16.
  #pragma unroll
  for (int tm = 0; tm < 2; ++tm) {
    #pragma unroll
    for (int tn = 0; tn < 2; ++tn) {
      const floatx8& acc = tm == 0 ? (tn == 0 ? acc00 : acc01)
                                   : (tn == 0 ? acc10 : acc11);
      const int gn = n0 + wn + tn * 16 + id;
      if (OUTMODE == 2) {
        // C^T[N,M]: this lane's 8 rows are contiguous -> one 16B store.
        const int gmb = m0 + wm + tm * 16 + 8 * half;
        unsigned int pk[4];
        #pragma unroll
        for (int r = 0; r < 4; ++r) {
          float v0 = acc[2 * r] * scale, v1 = acc[2 * r + 1] * scale;
          if (bias) { v0 += bias[gn]; v1 += bias[gn]; }
          if (RELU) { v0 = v0 > 0.f ? v0 : 0.f; v1 = v1 > 0.f ? v1 : 0.f; }
          pk[r] = (unsigned int)f32_to_bf16_bits(v0)
                | ((unsigned int)f32_to_bf16_bits(v1) << 16);
        }
        unsigned short* base = reinterpret_cast<unsigned short*>(Cv)
                             + (size_t)z * cStride + (size_t)gn * ldc + gmb;
        *reinterpret_cast<uint4*>(base) = *reinterpret_cast<uint4*>(pk);
      } else {
        #pragma unroll
        for (int r = 0; r < 8; ++r) {
          const int gm = m0 + wm + tm * 16 + r + 8 * half;
          float v = acc[r] * scale;
          if (bias) v += bias[gn];
          if (RELU) v = (v > 0.f) ? v : 0.f;
          size_t off = (size_t)z * cStride + (size_t)gm * ldc + gn;
          if (OUTMODE == 1) reinterpret_cast<unsigned short*>(Cv)[off] = f32_to_bf16_bits(v);
          else              reinterpret_cast<float*>(Cv)[off] = v;
        }
      }
    }
  }
}

// ---------------------------------------------------------------- pair bias
__global__ void pair_bias_kernel(const float* __restrict__ coords,
    const float* __restrict__ rot_w1,  const float* __restrict__ rot_b1,
    const float* __restrict__ rot_w2,  const float* __restrict__ rot_b2,
    const float* __restrict__ trans_w1,const float* __restrict__ trans_b1,
    const float* __restrict__ trans_w2,const float* __restrict__ trans_b2,
    const float* __restrict__ refl_w1, const float* __restrict__ refl_b1,
    const float* __restrict__ refl_w2, const float* __restrict__ refl_b2,
    float* __restrict__ biasOut)
{
  const long long total = (long long)BNUM * SNUM * SNUM;
  long long idx = (long long)blockIdx.x * blockDim.x + threadIdx.x;
  if (idx >= total) return;
  int j = (int)(idx % SNUM);
  int i = (int)((idx / SNUM) % SNUM);
  int b = (int)(idx / ((long long)SNUM * SNUM));

  float cix = coords[((size_t)b * SNUM + i) * 2 + 0];
  float ciy = coords[((size_t)b * SNUM + i) * 2 + 1];
  float cjx = coords[((size_t)b * SNUM + j) * 2 + 0];
  float cjy = coords[((size_t)b * SNUM + j) * 2 + 1];
  float dx = cjx - cix, dy = cjy - ciy;
  float dist = sqrtf(dx * dx + dy * dy + 1e-8f);
  float th = atan2f(dy, dx);
  float sn = sinf(th), cs = cosf(th);

  #pragma unroll
  for (int l = 0; l < LNUM; ++l) {
    float acc = 0.f;
    { // rot MLP: input [dist, sin, cos]
      const float* w1 = rot_w1 + (size_t)l * 3 * HNUM;
      const float* b1 = rot_b1 + (size_t)l * HNUM;
      const float* w2 = rot_w2 + (size_t)l * HNUM;
      for (int h = 0; h < HNUM; ++h) {
        float u = dist * w1[h] + sn * w1[HNUM + h] + cs * w1[2 * HNUM + h] + b1[h];
        acc += (u > 0.f ? u : 0.f) * w2[h];
      }
      acc += rot_b2[l];
    }
    { // trans MLP: input [dx, dy]
      const float* w1 = trans_w1 + (size_t)l * 2 * HNUM;
      const float* b1 = trans_b1 + (size_t)l * HNUM;
      const float* w2 = trans_w2 + (size_t)l * HNUM;
      for (int h = 0; h < HNUM; ++h) {
        float u = dx * w1[h] + dy * w1[HNUM + h] + b1[h];
        acc += (u > 0.f ? u : 0.f) * w2[h];
      }
      acc += trans_b2[l];
    }
    { // refl MLP: input [dx, dy, -dx, -dy]
      const float* w1 = refl_w1 + (size_t)l * 4 * HNUM;
      const float* b1 = refl_b1 + (size_t)l * HNUM;
      const float* w2 = refl_w2 + (size_t)l * HNUM;
      for (int h = 0; h < HNUM; ++h) {
        float u = dx * w1[h] + dy * w1[HNUM + h]
                - dx * w1[2 * HNUM + h] - dy * w1[3 * HNUM + h] + b1[h];
        acc += (u > 0.f ? u : 0.f) * w2[h];
      }
      acc += refl_b2[l];
    }
    biasOut[(size_t)l * total + idx] = acc;
  }
}

// ---------------------------------------------------------------- softmax
__global__ __launch_bounds__(256)
void softmax_bias_kernel(const float* __restrict__ scores,
                         const float* __restrict__ biasL,
                         unsigned short* __restrict__ attn)
{
  __shared__ float red[256];
  const int row = blockIdx.x;              // b*S + i
  const size_t base = (size_t)row * SNUM;
  const int t = threadIdx.x;
  float v[4]; float mx = -1e30f;
  #pragma unroll
  for (int q = 0; q < 4; ++q) {
    int c = t + q * 256;
    v[q] = scores[base + c] + biasL[base + c];
    mx = fmaxf(mx, v[q]);
  }
  red[t] = mx; __syncthreads();
  for (int s = 128; s > 0; s >>= 1) { if (t < s) red[t] = fmaxf(red[t], red[t + s]); __syncthreads(); }
  mx = red[0]; __syncthreads();
  float sum = 0.f;
  #pragma unroll
  for (int q = 0; q < 4; ++q) { v[q] = __expf(v[q] - mx); sum += v[q]; }
  red[t] = sum; __syncthreads();
  for (int s = 128; s > 0; s >>= 1) { if (t < s) red[t] += red[t + s]; __syncthreads(); }
  float inv = 1.f / red[0];
  #pragma unroll
  for (int q = 0; q < 4; ++q)
    attn[base + t + q * 256] = f32_to_bf16_bits(v[q] * inv);
}

// ---------------------------------------------------------------- residual + LN
__global__ __launch_bounds__(256)
void residual_ln_kernel(float* __restrict__ x, const float* __restrict__ delta,
                        const float* __restrict__ g, const float* __restrict__ be,
                        unsigned short* __restrict__ xb)
{
  __shared__ float red[256];
  const int row = blockIdx.x;
  const size_t base = (size_t)row * DNUM;
  const int t = threadIdx.x;
  float v0 = x[base + t]       + delta[base + t];
  float v1 = x[base + t + 256] + delta[base + t + 256];
  red[t] = v0 + v1; __syncthreads();
  for (int s = 128; s > 0; s >>= 1) { if (t < s) red[t] += red[t + s]; __syncthreads(); }
  float mu = red[0] * (1.f / DNUM); __syncthreads();
  float d0 = v0 - mu, d1 = v1 - mu;
  red[t] = d0 * d0 + d1 * d1; __syncthreads();
  for (int s = 128; s > 0; s >>= 1) { if (t < s) red[t] += red[t + s]; __syncthreads(); }
  float rstd = rsqrtf(red[0] * (1.f / DNUM) + 1e-5f);
  float o0 = d0 * rstd * g[t]       + be[t];
  float o1 = d1 * rstd * g[t + 256] + be[t + 256];
  x[base + t] = o0;           x[base + t + 256] = o1;
  xb[base + t] = f32_to_bf16_bits(o0);
  xb[base + t + 256] = f32_to_bf16_bits(o1);
}

// ---------------------------------------------------------------- final LN stats
__global__ __launch_bounds__(256)
void ln_stats_kernel(const float* __restrict__ x,
                     float* __restrict__ mu, float* __restrict__ rstd)
{
  __shared__ float red[256];
  const int row = blockIdx.x;
  const size_t base = (size_t)row * DNUM;
  const int t = threadIdx.x;
  float v0 = x[base + t], v1 = x[base + t + 256];
  red[t] = v0 + v1; __syncthreads();
  for (int s = 128; s > 0; s >>= 1) { if (t < s) red[t] += red[t + s]; __syncthreads(); }
  float m = red[0] * (1.f / DNUM); __syncthreads();
  float d0 = v0 - m, d1 = v1 - m;
  red[t] = d0 * d0 + d1 * d1; __syncthreads();
  for (int s = 128; s > 0; s >>= 1) { if (t < s) red[t] += red[t + s]; __syncthreads(); }
  if (t == 0) { mu[row] = m; rstd[row] = rsqrtf(red[0] * (1.f / DNUM) + 1e-5f); }
}

// ---------------------------------------------------------------- pool + fc
__global__ void pool_kernel(const float* __restrict__ x,
                            const float* __restrict__ mu, const float* __restrict__ rstd,
                            const float* __restrict__ g, const float* __restrict__ be,
                            float* __restrict__ pooled)
{
  int id = blockIdx.x * blockDim.x + threadIdx.x;
  if (id >= BNUM * DNUM) return;
  int b = id / DNUM, d = id % DNUM;
  float acc = 0.f;
  for (int i = 0; i < SNUM; ++i) {
    int row = b * SNUM + i;
    acc += (x[(size_t)row * DNUM + d] - mu[row]) * rstd[row] * g[d] + be[d];
  }
  pooled[id] = acc * (1.f / SNUM);
}

__global__ void fc_kernel(const float* __restrict__ pooled,
                          const float* __restrict__ w, const float* __restrict__ bias,
                          float* __restrict__ out)
{
  int id = blockIdx.x * blockDim.x + threadIdx.x;
  if (id >= BNUM * CNUM) return;
  int b = id / CNUM, c = id % CNUM;
  float acc = bias[c];
  for (int d = 0; d < DNUM; ++d)
    acc += pooled[b * DNUM + d] * w[(size_t)d * CNUM + c];
  out[id] = acc;
}

// ================================================================ launch
extern "C" void kernel_launch(void* const* d_in, const int* in_sizes, int n_in,
                              void* d_out, int out_size, void* d_ws, size_t ws_size,
                              hipStream_t stream) {
  const float* x_in     = (const float*)d_in[0];
  const float* coords   = (const float*)d_in[1];
  const float* Wq       = (const float*)d_in[2];
  const float* Wk       = (const float*)d_in[3];
  const float* Wv       = (const float*)d_in[4];
  const float* rot_w1   = (const float*)d_in[5];
  const float* rot_b1   = (const float*)d_in[6];
  const float* rot_w2   = (const float*)d_in[7];
  const float* rot_b2   = (const float*)d_in[8];
  const float* trans_w1 = (const float*)d_in[9];
  const float* trans_b1 = (const float*)d_in[10];
  const float* trans_w2 = (const float*)d_in[11];
  const float* trans_b2 = (const float*)d_in[12];
  const float* refl_w1  = (const float*)d_in[13];
  const float* refl_b1  = (const float*)d_in[14];
  const float* refl_w2  = (const float*)d_in[15];
  const float* refl_b2  = (const float*)d_in[16];
  const float* ln1_g    = (const float*)d_in[17];
  const float* ln1_b    = (const float*)d_in[18];
  const float* ffn_w1   = (const float*)d_in[19];
  const float* ffn_b1   = (const float*)d_in[20];
  const float* ffn_w2   = (const float*)d_in[21];
  const float* ffn_b2   = (const float*)d_in[22];
  const float* ln2_g    = (const float*)d_in[23];
  const float* ln2_b    = (const float*)d_in[24];
  const float* lnf_g    = (const float*)d_in[25];
  const float* lnf_b    = (const float*)d_in[26];
  const float* fc_w     = (const float*)d_in[27];
  const float* fc_b     = (const float*)d_in[28];

  const size_t nX  = (size_t)BNUM * SNUM * DNUM;   // 1M
  const size_t nSS = (size_t)BNUM * SNUM * SNUM;   // 2M
  const size_t nH  = (size_t)BNUM * SNUM * FNUM;   // 4M
  const size_t nWqkv = (size_t)LNUM * DNUM * DNUM;
  const size_t nWff  = (size_t)LNUM * DNUM * FNUM;

  char* ws = (char*)d_ws;
  size_t off = 0;
  auto alloc = [&](size_t bytes) -> char* {
    char* p = ws + off;
    off += (bytes + 255) & ~(size_t)255;
    return p;
  };
  float*          xf      = (float*)         alloc(nX * 4);
  unsigned short* xb      = (unsigned short*)alloc(nX * 2);
  unsigned short* wqT     = (unsigned short*)alloc(nWqkv * 2);  // [L][D][D] W^T
  unsigned short* wkT     = (unsigned short*)alloc(nWqkv * 2);
  unsigned short* wvT     = (unsigned short*)alloc(nWqkv * 2);
  unsigned short* w1T     = (unsigned short*)alloc(nWff * 2);   // [L][F][D]
  unsigned short* w2T     = (unsigned short*)alloc(nWff * 2);   // [L][D][F]
  unsigned short* qB      = (unsigned short*)alloc(nX * 2);     // [B*S][D]
  unsigned short* kB      = (unsigned short*)alloc(nX * 2);     // [B*S][D]
  unsigned short* vT      = (unsigned short*)alloc(nX * 2);     // [D][B*S]
  float*          scores  = (float*)         alloc(nSS * 4);
  float*          biasP   = (float*)         alloc((size_t)LNUM * nSS * 4);
  unsigned short* attnB   = (unsigned short*)alloc(nSS * 2);
  float*          attnOut = (float*)         alloc(nX * 4);
  unsigned short* hB      = (unsigned short*)alloc(nH * 2);
  float*          ffnOut  = (float*)         alloc(nX * 4);
  float*          muArr   = (float*)         alloc((size_t)BNUM * SNUM * 4);
  float*          rstdArr = (float*)         alloc((size_t)BNUM * SNUM * 4);
  float*          pooled  = (float*)         alloc((size_t)BNUM * DNUM * 4);
  (void)ws_size; (void)in_sizes; (void)n_in; (void)out_size;

  // x master copy (fp32)
  hipMemcpyAsync(xf, x_in, nX * 4, hipMemcpyDeviceToDevice, stream);

  // weights -> bf16, transposed to [N][K] for the K-major GEMM B path
  cvt_transpose_bf16_kernel<<<dim3(DNUM / 32, DNUM / 32, LNUM), 256, 0, stream>>>(Wq, wqT, DNUM, DNUM);
  cvt_transpose_bf16_kernel<<<dim3(DNUM / 32, DNUM / 32, LNUM), 256, 0, stream>>>(Wk, wkT, DNUM, DNUM);
  cvt_transpose_bf16_kernel<<<dim3(DNUM / 32, DNUM / 32, LNUM), 256, 0, stream>>>(Wv, wvT, DNUM, DNUM);
  cvt_transpose_bf16_kernel<<<dim3(FNUM / 32, DNUM / 32, LNUM), 256, 0, stream>>>(ffn_w1, w1T, DNUM, FNUM);
  cvt_transpose_bf16_kernel<<<dim3(DNUM / 32, FNUM / 32, LNUM), 256, 0, stream>>>(ffn_w2, w2T, FNUM, DNUM);

  // pairwise geometry bias, both layers
  pair_bias_kernel<<<(unsigned)(nSS / 256), 256, 0, stream>>>(
      coords, rot_w1, rot_b1, rot_w2, rot_b2,
      trans_w1, trans_b1, trans_w2, trans_b2,
      refl_w1, refl_b1, refl_w2, refl_b2, biasP);

  const float inv_sqrt_d = 0.04419417382415922f;  // 1/sqrt(512)
  const long long sSD = (long long)SNUM * DNUM;
  const long long sSS = (long long)SNUM * SNUM;
  const int MT = BNUM * SNUM;                     // 2048 flat rows

  for (int l = 0; l < LNUM; ++l) {
    cvt_f32_bf16_kernel<<<(unsigned)(nX / 256), 256, 0, stream>>>(xf, xb, (int)nX);

    // Q/K/V projections: [B*S,D] x [D,D]^T(of W^T) ; V written transposed.
    dim3 gProj(DNUM / 64, MT / 128, 1);
    gemm_wmma_kernel<1,0><<<gProj, 256, 0, stream>>>(
        xb, wqT + (size_t)l * DNUM * DNUM, qB, nullptr, 1.f,
        MT, DNUM, DNUM, DNUM, DNUM, DNUM, 0, 0, 0);
    gemm_wmma_kernel<1,0><<<gProj, 256, 0, stream>>>(
        xb, wkT + (size_t)l * DNUM * DNUM, kB, nullptr, 1.f,
        MT, DNUM, DNUM, DNUM, DNUM, DNUM, 0, 0, 0);
    gemm_wmma_kernel<2,0><<<gProj, 256, 0, stream>>>(        // vT: [D][B*S]
        xb, wvT + (size_t)l * DNUM * DNUM, vT, nullptr, 1.f,
        MT, DNUM, DNUM, DNUM, DNUM, MT /*ldc of C^T*/, 0, 0, 0);

    // scores = Q K^T / sqrt(D): B of GEMM is K in natural [S,D] layout.
    dim3 gScore(SNUM / 64, SNUM / 128, BNUM);
    gemm_wmma_kernel<0,0><<<gScore, 256, 0, stream>>>(
        qB, kB, scores, nullptr, inv_sqrt_d,
        SNUM, SNUM, DNUM, DNUM, DNUM, SNUM, sSD, sSD, sSS);

    // softmax(scores + bias[l]) -> attn bf16
    softmax_bias_kernel<<<BNUM * SNUM, 256, 0, stream>>>(
        scores, biasP + (size_t)l * nSS, attnB);

    // attn @ V: B = vT [D][B*S] (K-major), batch column offset b*S.
    dim3 gAV(DNUM / 64, SNUM / 128, BNUM);
    gemm_wmma_kernel<0,0><<<gAV, 256, 0, stream>>>(
        attnB, vT, attnOut, nullptr, 1.f,
        SNUM, DNUM, SNUM, SNUM, MT /*ldb*/, DNUM,
        sSS, (long long)SNUM /*bStride*/, sSD);

    // x = LN(x + attn_out)
    residual_ln_kernel<<<BNUM * SNUM, 256, 0, stream>>>(
        xf, attnOut, ln1_g + (size_t)l * DNUM, ln1_b + (size_t)l * DNUM, xb);

    // FFN
    dim3 gF1(FNUM / 64, MT / 128, 1);
    gemm_wmma_kernel<1,1><<<gF1, 256, 0, stream>>>(
        xb, w1T + (size_t)l * DNUM * FNUM, hB, ffn_b1 + (size_t)l * FNUM, 1.f,
        MT, FNUM, DNUM, DNUM, DNUM, FNUM, 0, 0, 0);
    dim3 gF2(DNUM / 64, MT / 128, 1);
    gemm_wmma_kernel<0,0><<<gF2, 256, 0, stream>>>(
        hB, w2T + (size_t)l * FNUM * DNUM, ffnOut, ffn_b2 + (size_t)l * DNUM, 1.f,
        MT, DNUM, FNUM, FNUM, FNUM, DNUM, 0, 0, 0);

    // x = LN(x + ffn)
    residual_ln_kernel<<<BNUM * SNUM, 256, 0, stream>>>(
        xf, ffnOut, ln2_g + (size_t)l * DNUM, ln2_b + (size_t)l * DNUM, xb);
  }

  // final LN -> mean pool -> fc
  ln_stats_kernel<<<BNUM * SNUM, 256, 0, stream>>>(xf, muArr, rstdArr);
  pool_kernel<<<(BNUM * DNUM + 255) / 256, 256, 0, stream>>>(
      xf, muArr, rstdArr, lnf_g, lnf_b, pooled);
  fc_kernel<<<(BNUM * CNUM + 255) / 256, 256, 0, stream>>>(
      pooled, fc_w, fc_b, (float*)d_out);
}
